// LinearCausalAttention_11304353923655
// MI455X (gfx1250) — matrix operand order repaired and checked by reading, not compiled, not run on hardware
//
#include <hip/hip_runtime.h>
#include <hip/hip_bf16.h>
#include <math.h>

typedef __attribute__((ext_vector_type(16))) _Float16 v16h;
typedef __attribute__((ext_vector_type(8)))  _Float16 v8h;
typedef __attribute__((ext_vector_type(8)))  float    v8f;

#define WMMA_F16(a,b,c) __builtin_amdgcn_wmma_f32_16x16x32_f16( \
    false, (a), false, (b), (short)0, (c), false, false)

constexpr int B_ = 2, T_ = 2048, C_ = 1024, H_ = 16, D_ = 64;
constexpr int NT    = B_ * T_;      // 4096 tokens
constexpr int CHUNK = 32;           // == WMMA K for f16
constexpr int NC    = T_ / CHUNK;   // 64 chunks per (b,h)

// ---- workspace layout (bytes) ----
constexpr size_t OFF_XH  = 0;                                      // NT*C f16
constexpr size_t OFF_WH  = OFF_XH  + (size_t)NT * C_ * 2;          // 4*C*C f16 (Wq,Wk,Wv,Wp)
constexpr size_t OFF_QP  = OFF_WH  + (size_t)4 * C_ * C_ * 2;      // B*H*T*D f16
constexpr size_t OFF_KP  = OFF_QP  + (size_t)B_ * H_ * T_ * D_ * 2;
constexpr size_t OFF_VH  = OFF_KP  + (size_t)B_ * H_ * T_ * D_ * 2;
constexpr size_t OFF_YH  = OFF_VH  + (size_t)B_ * H_ * T_ * D_ * 2; // NT*C f16
constexpr size_t OFF_KVC = OFF_YH  + (size_t)NT * C_ * 2;           // B*H*NC*D*D f32
constexpr size_t OFF_KVP = OFF_KVC + (size_t)B_ * H_ * NC * D_ * D_ * 4; // f16 prefix
constexpr size_t OFF_KSC = OFF_KVP + (size_t)B_ * H_ * NC * D_ * D_ * 2; // f32
constexpr size_t OFF_KSP = OFF_KSC + (size_t)B_ * H_ * NC * D_ * 4;      // f32

constexpr int OUT_KV = NT * C_;
constexpr int OUT_KS = OUT_KV + B_ * H_ * D_ * D_;

// -------- operand builders (contiguous k order, same for A and B) --------

// A-matrix 16x32 f16: lane L holds row M; halves j -> K = k0 + (L>>4)*16 + j
__device__ __forceinline__ v16h load_a_rowmajor(const _Float16* __restrict__ row,
                                                int k0, int g) {
  const v8h lo = *(const v8h*)(row + k0 + g * 16);
  const v8h hi = *(const v8h*)(row + k0 + g * 16 + 8);
  v16h a;
#pragma unroll
  for (int i = 0; i < 8; i++) { a[i] = lo[i]; a[i + 8] = hi[i]; }
  return a;
}

// B-matrix 32x16 f16: lane L holds column N; halves j -> K = k0 + (L>>4)*16 + j
__device__ __forceinline__ v16h load_b_contig(const _Float16* __restrict__ col,
                                              int k0, int g) {
  const v8h lo = *(const v8h*)(col + k0 + g * 16);
  const v8h hi = *(const v8h*)(col + k0 + g * 16 + 8);
  v16h b;
#pragma unroll
  for (int i = 0; i < 8; i++) { b[i] = lo[i]; b[i + 8] = hi[i]; }
  return b;
}

__device__ __forceinline__ v16h gather_b_strided(const _Float16* __restrict__ base,
                                                 int stride, int g) {
  v16h b;
#pragma unroll
  for (int j = 0; j < 16; j++) b[j] = base[(g * 16 + j) * stride];
  return b;
}

__device__ __forceinline__ v16h gather_a_strided(const _Float16* __restrict__ base,
                                                 int stride, int g) {
  v16h a;
#pragma unroll
  for (int j = 0; j < 8; j++) a[j] = base[(g * 16 + j) * stride];
#pragma unroll
  for (int j = 0; j < 8; j++) a[j + 8] = base[(g * 16 + 8 + j) * stride];
  return a;
}

// -------- CDNA5 async copy: global -> LDS, 16B per lane --------
__device__ __forceinline__ void async_copy_16B(unsigned lds_addr, unsigned gvoff,
                                               unsigned long long sbase) {
  asm volatile("global_load_async_to_lds_b128 %0, %1, %2"
               :: "v"(lds_addr), "v"(gvoff), "s"(sbase)
               : "memory");
}

// -------- kernel 0: f32 -> f16 convert --------
__global__ void f32_to_f16_kernel(const float* __restrict__ src,
                                  _Float16* __restrict__ dst, int n) {
  int i = blockIdx.x * 256 + threadIdx.x;
  int stride = gridDim.x * 256;
  for (; i < n; i += stride) dst[i] = (_Float16)src[i];
}

// ---------------------------------------------------------------------------
// Shared GEMM mainloop: block (256 thr / 8 waves) computes a 256(M) x 64(N)
// tile of A @ W^T, K = C_ = 1024. Double-buffered LDS staging of 64-wide
// k-slices via global_load_async_to_lds_b128 (10 async issues / thread /
// slice). Each wave owns two 16-row m-tiles -> 16 WMMAs per barrier pair,
// every staged B operand feeds 2 WMMAs.
// ---------------------------------------------------------------------------
constexpr int MBLK  = 256;                 // block M tile
constexpr int KSTEP = 64;                  // halves per staged k-slice
constexpr int NK    = C_ / KSTEP;          // 16 slices
constexpr int ABUF_B = MBLK * KSTEP * 2;   // 32768 bytes per A buffer
constexpr int BBUF_B = 64 * KSTEP * 2;     // 8192 bytes per B buffer

__device__ __forceinline__ void gemm_mainloop_256x64(
    const _Float16* __restrict__ Aglob, const _Float16* __restrict__ Bglob,
    int mbase, int nbase, _Float16* Ab, _Float16* Bb, v8f acc[2][4]) {
  const int tid  = threadIdx.x;
  const int lane = tid & 31, g = lane >> 4, ln = lane & 15;
  const int wv   = tid >> 5;

  // per-thread async-copy coordinates
  // A: thread tid owns full 128B row-slice of row (mbase+tid): 8 x 16B
  const unsigned a_voff0 = (unsigned)(mbase + tid) * (C_ * 2);
  // B: thread owns 32B of row (nbase + tid>>2): 2 x 16B
  const int brow = tid >> 2;
  const int bcol = (tid & 3) * 32;
  const unsigned b_voff0 = (unsigned)(nbase + brow) * (C_ * 2) + (unsigned)bcol;

  const unsigned a_lds0 = (unsigned)(size_t)(void*)Ab + (unsigned)(tid * 128);
  const unsigned b_lds0 = (unsigned)(size_t)(void*)Bb + (unsigned)(brow * 128 + bcol);
  const unsigned long long abase = (unsigned long long)(size_t)Aglob;
  const unsigned long long bbase = (unsigned long long)(size_t)Bglob;

  // issue k-slice `s` into buffer parity p (10 async instructions / thread)
  auto issue = [&](int s, int p) {
    const unsigned kb = (unsigned)(s * KSTEP * 2);
#pragma unroll
    for (int j = 0; j < 8; j++)
      async_copy_16B(a_lds0 + p * ABUF_B + j * 16, a_voff0 + kb + j * 16, abase);
    async_copy_16B(b_lds0 + p * BBUF_B,      b_voff0 + kb,      bbase);
    async_copy_16B(b_lds0 + p * BBUF_B + 16, b_voff0 + kb + 16, bbase);
  };

  issue(0, 0);
  issue(1, 1);

  for (int i = 0; i < NK; i++) {
    if (i + 1 < NK) asm volatile("s_wait_asynccnt 0xa" ::: "memory");
    else            asm volatile("s_wait_asynccnt 0x0" ::: "memory");
    __syncthreads();                         // slice i resident for all waves

    const int p = i & 1;
    const _Float16* Abuf = Ab + p * (MBLK * KSTEP);
    const _Float16* Bbuf = Bb + p * (64 * KSTEP);
    const _Float16* ar0 = Abuf + (wv * 32 + ln) * KSTEP;
    const _Float16* ar1 = Abuf + (wv * 32 + 16 + ln) * KSTEP;
#pragma unroll
    for (int kk = 0; kk < KSTEP; kk += 32) {
      v16h a0 = load_a_rowmajor(ar0, kk, g);
      v16h a1 = load_a_rowmajor(ar1, kk, g);
#pragma unroll
      for (int nsub = 0; nsub < 4; nsub++) {
        v16h bm = load_b_contig(Bbuf + (nsub * 16 + ln) * KSTEP, kk, g);
        acc[0][nsub] = WMMA_F16(a0, bm, acc[0][nsub]);
        acc[1][nsub] = WMMA_F16(a1, bm, acc[1][nsub]);
      }
    }
    __syncthreads();                         // all waves done reading buffer p
    if (i + 2 < NK) issue(i + 2, p);         // refill the freed buffer
  }
}

// -------- kernel 1: QKV GEMM (out = x @ W.T), feature map on q,k --------
// grid: 3 (q/k/v) x 16 (M/256) x 16 (N/64) blocks of 256 threads
__global__ void __launch_bounds__(256) qkv_gemm_kernel(
    const _Float16* __restrict__ xh, const _Float16* __restrict__ wh,
    _Float16* __restrict__ qp, _Float16* __restrict__ kp,
    _Float16* __restrict__ vh) {
  __shared__ __align__(16) _Float16 Ab[2 * MBLK * KSTEP];
  __shared__ __align__(16) _Float16 Bb[2 * 64 * KSTEP];
  const int tid  = threadIdx.x;
  const int lane = tid & 31, g = lane >> 4, ln = lane & 15;
  const int wv   = tid >> 5;

  const int blk  = blockIdx.x;
  const int wsel = blk >> 8;                 // / 256
  const int rem  = blk & 255;
  const int mbase = (rem >> 4) * MBLK;
  const int h     = rem & 15;
  const int nbase = h * 64;

  v8f acc[2][4] = {};
  gemm_mainloop_256x64(xh, wh + (size_t)wsel * C_ * C_, mbase, nbase, Ab, Bb, acc);

  _Float16* out = (wsel == 0) ? qp : (wsel == 1 ? kp : vh);
  const bool fmap = (wsel < 2);
#pragma unroll
  for (int mt = 0; mt < 2; mt++) {
    const int m0 = mbase + wv * 32 + mt * 16;
#pragma unroll
    for (int nsub = 0; nsub < 4; nsub++) {
      const int d = nsub * 16 + ln;
#pragma unroll
      for (int r = 0; r < 8; r++) {
        const int tok = m0 + r + 8 * g;
        const int b   = tok >> 11;
        const int t   = tok & 2047;
        float v = acc[mt][nsub][r];
        if (fmap) v = (v > 0.f) ? (v + 1.f) : __expf(v);
        out[(size_t)((b * H_ + h) * T_ + t) * D_ + d] = (_Float16)v;
      }
    }
  }
}

// -------- kernel 5: output projection (out = y @ Wp.T + bp), fp32 out --------
__global__ void __launch_bounds__(256) proj_gemm_kernel(
    const _Float16* __restrict__ yh, const _Float16* __restrict__ wph,
    const float* __restrict__ bp, float* __restrict__ out) {
  __shared__ __align__(16) _Float16 Ab[2 * MBLK * KSTEP];
  __shared__ __align__(16) _Float16 Bb[2 * 64 * KSTEP];
  const int tid  = threadIdx.x;
  const int lane = tid & 31, g = lane >> 4, ln = lane & 15;
  const int wv   = tid >> 5;

  const int blk   = blockIdx.x;              // 256 blocks
  const int mbase = (blk >> 4) * MBLK;
  const int nbase = (blk & 15) * 64;

  v8f acc[2][4] = {};
  gemm_mainloop_256x64(yh, wph, mbase, nbase, Ab, Bb, acc);

#pragma unroll
  for (int mt = 0; mt < 2; mt++) {
    const int m0 = mbase + wv * 32 + mt * 16;
#pragma unroll
    for (int nsub = 0; nsub < 4; nsub++) {
      const int n = nbase + nsub * 16 + ln;
      const float bias = bp[n];
#pragma unroll
      for (int r = 0; r < 8; r++) {
        const int tok = m0 + r + 8 * g;
        out[(size_t)tok * C_ + n] = acc[mt][nsub][r] + bias;
      }
    }
  }
}

// -------- kernel 2 (pass A): per-chunk states KV_c = Kc^T @ Vc, ksum_c --------
__global__ void __launch_bounds__(32) chunk_state_kernel(
    const _Float16* __restrict__ kp, const _Float16* __restrict__ vh,
    float* __restrict__ kvc, float* __restrict__ ksc) {
  const int lane = threadIdx.x & 31, g = lane >> 4, ln = lane & 15;
  const int wid = blockIdx.x;
  const int c  = wid & (NC - 1);
  const int bh = wid >> 6;
  const _Float16* kc = kp + ((size_t)bh * T_ + c * CHUNK) * D_;
  const _Float16* vc = vh + ((size_t)bh * T_ + c * CHUNK) * D_;

  v8f acc[4][4] = {};
  v16h Bt[4];
#pragma unroll
  for (int nt = 0; nt < 4; nt++)
    Bt[nt] = gather_b_strided(vc + nt * 16 + ln, D_, g);
#pragma unroll
  for (int mt = 0; mt < 4; mt++) {
    v16h A = gather_a_strided(kc + mt * 16 + ln, D_, g);
#pragma unroll
    for (int nt = 0; nt < 4; nt++) acc[mt][nt] = WMMA_F16(A, Bt[nt], acc[mt][nt]);
  }

  float* kvbase = kvc + ((size_t)bh * NC + c) * (D_ * D_);
#pragma unroll
  for (int mt = 0; mt < 4; mt++)
#pragma unroll
    for (int nt = 0; nt < 4; nt++)
#pragma unroll
      for (int r = 0; r < 8; r++) {
        const int d = mt * 16 + r + 8 * g, dv = nt * 16 + ln;
        kvbase[d * 64 + dv] = acc[mt][nt][r];
      }

  float s0 = 0.f, s1 = 0.f;
  const int d0 = lane * 2;
  for (int t = 0; t < CHUNK; t++) {
    s0 += (float)kc[t * D_ + d0];
    s1 += (float)kc[t * D_ + d0 + 1];
  }
  float* ksbase = ksc + ((size_t)bh * NC + c) * D_;
  ksbase[d0] = s0; ksbase[d0 + 1] = s1;
}

// -------- kernel 3 (pass B): exclusive prefix scan over chunks --------
__global__ void __launch_bounds__(256) prefix_scan_kernel(
    const float* __restrict__ kvc, const float* __restrict__ ksc,
    _Float16* __restrict__ kvp, float* __restrict__ ksp,
    float* __restrict__ out_kv, float* __restrict__ out_ks) {
  const int bh = blockIdx.x;
  const int tid = threadIdx.x;
  for (int e = tid; e < D_ * D_; e += 256) {
    float run = 0.f;
    for (int c = 0; c < NC; c++) {
      const size_t idx = ((size_t)bh * NC + c) * (D_ * D_) + e;
      kvp[idx] = (_Float16)run;
      run += kvc[idx];
    }
    out_kv[(size_t)bh * (D_ * D_) + e] = run;
  }
  if (tid < D_) {
    float run = 0.f;
    for (int c = 0; c < NC; c++) {
      const size_t idx = ((size_t)bh * NC + c) * D_ + tid;
      ksp[idx] = run;
      run += ksc[idx];
    }
    out_ks[bh * D_ + tid] = run;
  }
}

// -------- kernel 4 (pass C): per-chunk output --------
__global__ void __launch_bounds__(32) chunk_out_kernel(
    const _Float16* __restrict__ qp, const _Float16* __restrict__ kp,
    const _Float16* __restrict__ vh, const _Float16* __restrict__ kvp,
    const float* __restrict__ ksp, _Float16* __restrict__ yh) {
  __shared__ __align__(32) _Float16 Sld[CHUNK * CHUNK];
  const int lane = threadIdx.x & 31, g = lane >> 4, ln = lane & 15;
  const int wid = blockIdx.x;
  const int c  = wid & (NC - 1);
  const int bh = wid >> 6;
  const int h  = bh & (H_ - 1);
  const int b  = bh >> 4;
  const _Float16* qc   = qp + ((size_t)bh * T_ + c * CHUNK) * D_;
  const _Float16* kc   = kp + ((size_t)bh * T_ + c * CHUNK) * D_;
  const _Float16* vc   = vh + ((size_t)bh * T_ + c * CHUNK) * D_;
  const _Float16* kvpc = kvp + ((size_t)bh * NC + c) * (D_ * D_);
  const float*    kspc = ksp + ((size_t)bh * NC + c) * D_;

  // S = Qc @ Kc^T (32x32)
  v8f s[2][2] = {};
#pragma unroll
  for (int ks = 0; ks < 2; ks++) {
    const int k0 = ks * 32;
    v16h A[2];
#pragma unroll
    for (int mt = 0; mt < 2; mt++)
      A[mt] = load_a_rowmajor(qc + (size_t)(mt * 16 + ln) * D_, k0, g);
#pragma unroll
    for (int nt = 0; nt < 2; nt++) {
      v16h Bm = load_b_contig(kc + (size_t)(nt * 16 + ln) * D_, k0, g);
#pragma unroll
      for (int mt = 0; mt < 2; mt++) s[mt][nt] = WMMA_F16(A[mt], Bm, s[mt][nt]);
    }
  }

  float rowsum[16];
#pragma unroll
  for (int mt = 0; mt < 2; mt++)
#pragma unroll
    for (int r = 0; r < 8; r++) {
      const int trow = mt * 16 + r + 8 * g;
      float rs = 0.f;
#pragma unroll
      for (int nt = 0; nt < 2; nt++) {
        const int tc = nt * 16 + ln;
        const float val = (tc <= trow) ? s[mt][nt][r] : 0.f;
        rs += val;
        Sld[trow * CHUNK + tc] = (_Float16)val;
      }
      rowsum[mt * 8 + r] = rs;
    }
#pragma unroll
  for (int i = 0; i < 16; i++) {
    float v = rowsum[i];
    v += __shfl_xor(v, 1);
    v += __shfl_xor(v, 2);
    v += __shfl_xor(v, 4);
    v += __shfl_xor(v, 8);
    rowsum[i] = v;
  }
  __syncthreads();

  float den0 = 1e-6f;
#pragma unroll 8
  for (int d = 0; d < D_; d++) den0 += (float)qc[lane * D_ + d] * kspc[d];

  v8f num[2][4] = {};
  {
    v16h As[2];
#pragma unroll
    for (int mt = 0; mt < 2; mt++)
      As[mt] = load_a_rowmajor(&Sld[(mt * 16 + ln) * CHUNK], 0, g);
#pragma unroll
    for (int nvt = 0; nvt < 4; nvt++) {
      v16h Bm = gather_b_strided(vc + nvt * 16 + ln, D_, g);
#pragma unroll
      for (int mt = 0; mt < 2; mt++) num[mt][nvt] = WMMA_F16(As[mt], Bm, num[mt][nvt]);
    }
  }
#pragma unroll
  for (int ks = 0; ks < 2; ks++) {
    const int k0 = ks * 32;
    v16h Aq[2];
#pragma unroll
    for (int mt = 0; mt < 2; mt++)
      Aq[mt] = load_a_rowmajor(qc + (size_t)(mt * 16 + ln) * D_, k0, g);
#pragma unroll
    for (int nvt = 0; nvt < 4; nvt++) {
      v16h Bm = gather_b_strided(kvpc + (size_t)k0 * D_ + nvt * 16 + ln, D_, g);
#pragma unroll
      for (int mt = 0; mt < 2; mt++) num[mt][nvt] = WMMA_F16(Aq[mt], Bm, num[mt][nvt]);
    }
  }

  _Float16* yrow = yh + ((size_t)(b * T_ + c * CHUNK)) * C_ + h * D_;
#pragma unroll
  for (int mt = 0; mt < 2; mt++)
#pragma unroll
    for (int r = 0; r < 8; r++) {
      const int trow = mt * 16 + r + 8 * g;
      const float den = __shfl(den0, trow) + rowsum[mt * 8 + r];
#pragma unroll
      for (int nvt = 0; nvt < 4; nvt++) {
        const int dv = nvt * 16 + ln;
        yrow[(size_t)trow * C_ + dv] = (_Float16)(num[mt][nvt][r] / den);
      }
    }
}

extern "C" void kernel_launch(void* const* d_in, const int* in_sizes, int n_in,
                              void* d_out, int out_size, void* d_ws, size_t ws_size,
                              hipStream_t stream) {
  const float* x  = (const float*)d_in[0];
  const float* Wq = (const float*)d_in[1];
  const float* Wk = (const float*)d_in[2];
  const float* Wv = (const float*)d_in[3];
  const float* Wp = (const float*)d_in[4];
  const float* bp = (const float*)d_in[5];

  char* ws = (char*)d_ws;
  _Float16* xh  = (_Float16*)(ws + OFF_XH);
  _Float16* wh  = (_Float16*)(ws + OFF_WH);
  _Float16* qp  = (_Float16*)(ws + OFF_QP);
  _Float16* kp  = (_Float16*)(ws + OFF_KP);
  _Float16* vh  = (_Float16*)(ws + OFF_VH);
  _Float16* yh  = (_Float16*)(ws + OFF_YH);
  float*    kvc = (float*)(ws + OFF_KVC);
  _Float16* kvp = (_Float16*)(ws + OFF_KVP);
  float*    ksc = (float*)(ws + OFF_KSC);
  float*    ksp = (float*)(ws + OFF_KSP);
  float*    out = (float*)d_out;

  f32_to_f16_kernel<<<512, 256, 0, stream>>>(x, xh, NT * C_);
  f32_to_f16_kernel<<<256, 256, 0, stream>>>(Wq, wh + (size_t)0 * C_ * C_, C_ * C_);
  f32_to_f16_kernel<<<256, 256, 0, stream>>>(Wk, wh + (size_t)1 * C_ * C_, C_ * C_);
  f32_to_f16_kernel<<<256, 256, 0, stream>>>(Wv, wh + (size_t)2 * C_ * C_, C_ * C_);
  f32_to_f16_kernel<<<256, 256, 0, stream>>>(Wp, wh + (size_t)3 * C_ * C_, C_ * C_);

  qkv_gemm_kernel<<<3 * 16 * 16, 256, 0, stream>>>(xh, wh, qp, kp, vh);
  chunk_state_kernel<<<B_ * H_ * NC, 32, 0, stream>>>(kp, vh, kvc, ksc);
  prefix_scan_kernel<<<B_ * H_, 256, 0, stream>>>(kvc, ksc, kvp, ksp,
                                                  out + OUT_KV, out + OUT_KS);
  chunk_out_kernel<<<B_ * H_ * NC, 32, 0, stream>>>(qp, kp, vh, kvp, ksp, yh);
  proj_gemm_kernel<<<16 * 16, 256, 0, stream>>>(yh, wh + (size_t)3 * C_ * C_, bp, out);
}
